// Attention_53944789238074
// MI455X (gfx1250) — matrix operand order repaired and checked by reading, not compiled, not run on hardware
//
#include <hip/hip_runtime.h>

typedef __bf16 bf16;
typedef __attribute__((ext_vector_type(16))) __bf16 v16bf;
typedef __attribute__((ext_vector_type(8)))  __bf16 v8bf;
typedef __attribute__((ext_vector_type(8)))  float  v8f;

#define D_MODEL      2048
#define NUM_HEADS    32
#define NUM_KV_HEADS 8
#define HEAD_DIM     64
#define BATCH        2
#define SEQ          2048
#define TOKENS       (BATCH*SEQ)   /* 4096 */
#define QKV_OUT      3072

#define CAT16(lo,hi) __builtin_shufflevector(lo,hi,0,1,2,3,4,5,6,7,8,9,10,11,12,13,14,15)

#if defined(__AMDGCN__) && __has_builtin(__builtin_amdgcn_global_load_async_to_lds_b128) && __has_builtin(__builtin_amdgcn_s_wait_asynccnt)
#define HAVE_ASYNC_LDS 1
#else
#define HAVE_ASYNC_LDS 0
#endif

// 16-byte global -> LDS copy (async DMA on gfx1250 when available)
__device__ __forceinline__ void cp16(const bf16* g, bf16* l) {
#if HAVE_ASYNC_LDS
  typedef int v4i __attribute__((vector_size(16)));
  typedef __attribute__((address_space(1))) v4i gv4i;
  typedef __attribute__((address_space(3))) v4i lv4i;
  __builtin_amdgcn_global_load_async_to_lds_b128((gv4i*)g, (lv4i*)l, 0, 0);
#else
  *(v8bf*)l = *(const v8bf*)g;
#endif
}

template <int N>
__device__ __forceinline__ void wait_async_le() {
#if HAVE_ASYNC_LDS
  __builtin_amdgcn_s_wait_asynccnt(N);
#endif
}

// ---------------------------------------------------------------- converts
__global__ void k_f32_to_bf16(const float* __restrict__ x, bf16* __restrict__ y, int n) {
  int i = blockIdx.x * blockDim.x + threadIdx.x;
  if (i < n) y[i] = (bf16)x[i];
}

// w: [K][N] fp32 -> wT: [N][K] bf16
__global__ void k_transpose_bf16(const float* __restrict__ w, bf16* __restrict__ wT, int K, int N) {
  int i = blockIdx.x * blockDim.x + threadIdx.x;
  if (i >= K * N) return;
  int n = i / K, k = i % K;
  wT[i] = (bf16)w[(size_t)k * N + n];
}

// ---------------------------------------------------------------- GEMM (bf16 WMMA, f32 accum)
// C[M][N] = A[M][K] @ BT[N][K]^T + bias.
// Block: 8 waves -> 128(M) x 128(N) tile; wave (wm,wn): 32(M) x 64(N) = 8 accumulators.
// A/B panels (128x32 bf16 each) double-buffered in LDS, filled via async global->LDS DMA.
__global__ __launch_bounds__(256)
void k_gemm_bf16(const bf16* __restrict__ A, const bf16* __restrict__ BT,
                 const float* __restrict__ bias, float* __restrict__ C,
                 int M, int N, int K) {
  __shared__ __align__(16) bf16 sA[2][128 * 32];
  __shared__ __align__(16) bf16 sB[2][128 * 32];

  const int tid  = threadIdx.x;
  const int lane = tid & 31;
  const int wave = tid >> 5;
  const int wm   = wave & 3;         // 4 waves along M (32 rows each)
  const int wn   = wave >> 2;        // 2 waves along N (64 cols each)
  const int m0b  = blockIdx.y * 128;
  const int n0b  = blockIdx.x * 128;
  const int mrow = lane & 15;
  const int kb   = (lane < 16) ? 0 : 8;    // A-layout K base (elements)
  const int kb2  = (lane < 16) ? 0 : 16;   // unused; B tiles load with kb

  // copy assignment: 256 threads, 128 rows x 64B per panel -> 2 threads/row, 32B each
  const int crow  = tid >> 1;
  const int chalf = (tid & 1) * 16;        // element offset within 32-elem row
  const bf16* gA = A  + (size_t)(m0b + crow) * K + chalf;
  const bf16* gB = BT + (size_t)(n0b + crow) * K + chalf;

  v8f acc[2][4] = {};

  const int nsteps = K / 32;

  // prologue: fill buffer 0
  {
    bf16* la = &sA[0][crow * 32 + chalf];
    bf16* lb = &sB[0][crow * 32 + chalf];
    cp16(gA, la); cp16(gA + 8, la + 8);
    cp16(gB, lb); cp16(gB + 8, lb + 8);
  }

  for (int s = 0; s < nsteps; s++) {
    if (s + 1 < nsteps) {
      const int k0 = (s + 1) * 32;
      bf16* la = &sA[(s + 1) & 1][crow * 32 + chalf];
      bf16* lb = &sB[(s + 1) & 1][crow * 32 + chalf];
      cp16(gA + k0, la); cp16(gA + k0 + 8, la + 8);
      cp16(gB + k0, lb); cp16(gB + k0 + 8, lb + 8);
      if (k0 + 32 < K) {
        __builtin_prefetch(gA + k0 + 32, 0, 1);
        __builtin_prefetch(gB + k0 + 32, 0, 1);
      }
      wait_async_le<4>();   // previous panel complete; next panel still in flight
    } else {
      wait_async_le<0>();
    }
    __syncthreads();

    const bf16* pa = &sA[s & 1][(wm * 32) * 32];
    const bf16* pb = &sB[s & 1][(wn * 64) * 32];

    v16bf a0, a1;
    { const bf16* p = pa + mrow * 32 + kb;
      v8bf lo = *(const v8bf*)p, hi = *(const v8bf*)(p + 16); a0 = CAT16(lo, hi); }
    { const bf16* p = pa + (16 + mrow) * 32 + kb;
      v8bf lo = *(const v8bf*)p, hi = *(const v8bf*)(p + 16); a1 = CAT16(lo, hi); }
#pragma unroll
    for (int nt = 0; nt < 4; nt++) {
      const bf16* pbr = pb + (nt * 16 + mrow) * 32 + kb;
      v16bf b = CAT16(*(const v8bf*)pbr, *(const v8bf*)(pbr + 16));
      acc[0][nt] = __builtin_amdgcn_wmma_f32_16x16x32_bf16(false, a0, false, b,
                                                           (short)0, acc[0][nt], false, false);
      acc[1][nt] = __builtin_amdgcn_wmma_f32_16x16x32_bf16(false, a1, false, b,
                                                           (short)0, acc[1][nt], false, false);
    }
    asm volatile("s_wait_dscnt 0" ::: "memory");  // LDS reads done before DMA overwrites buffer
    __syncthreads();
  }

#pragma unroll
  for (int g = 0; g < 2; g++) {
#pragma unroll
    for (int nt = 0; nt < 4; nt++) {
      int n = n0b + wn * 64 + nt * 16 + mrow;
      float bv = bias[n];
#pragma unroll
      for (int v = 0; v < 8; v++) {
        int r = m0b + wm * 32 + g * 16 + v + ((lane < 16) ? 0 : 8);
        C[(size_t)r * N + n] = acc[g][nt][v] + bv;
      }
    }
  }
}

// ---------------------------------------------------------------- RoPE + pack
// ln(10000)/32
#define ROPE_C 0.28782313662425572f

__global__ void k_rope_q(const float* __restrict__ qkv, bf16* __restrict__ Q) {
  int i = blockIdx.x * blockDim.x + threadIdx.x;       // TOKENS*32 heads*32 pairs
  if (i >= TOKENS * NUM_HEADS * 32) return;
  int dp = i & 31; int t = i >> 5;
  int h = t & (NUM_HEADS - 1); int tok = t >> 5;
  int b = tok / SEQ, s = tok % SEQ;
  float inv = __expf(-(float)dp * ROPE_C);
  float sn, cs; __sincosf((float)s * inv, &sn, &cs);
  const float* src = qkv + (size_t)tok * QKV_OUT + h * HEAD_DIM + 2 * dp;
  float x1 = src[0], x2 = src[1];
  size_t dst = (((size_t)(b * NUM_HEADS + h)) * SEQ + s) * HEAD_DIM + 2 * dp;
  Q[dst]     = (bf16)(x1 * cs - x2 * sn);
  Q[dst + 1] = (bf16)(x1 * sn + x2 * cs);
}

__global__ void k_rope_k(const float* __restrict__ qkv, bf16* __restrict__ Kp) {
  int i = blockIdx.x * blockDim.x + threadIdx.x;       // TOKENS*8 kvheads*32 pairs
  if (i >= TOKENS * NUM_KV_HEADS * 32) return;
  int dp = i & 31; int t = i >> 5;
  int kvh = t & (NUM_KV_HEADS - 1); int tok = t >> 3;
  int b = tok / SEQ, s = tok % SEQ;
  float inv = __expf(-(float)dp * ROPE_C);
  float sn, cs; __sincosf((float)s * inv, &sn, &cs);
  const float* src = qkv + (size_t)tok * QKV_OUT + D_MODEL + kvh * HEAD_DIM + 2 * dp;
  float x1 = src[0], x2 = src[1];
  size_t dst = (((size_t)(b * NUM_KV_HEADS + kvh)) * SEQ + s) * HEAD_DIM + 2 * dp;
  Kp[dst]     = (bf16)(x1 * cs - x2 * sn);
  Kp[dst + 1] = (bf16)(x1 * sn + x2 * cs);
}

// V stored transposed: Vt[b][kvh][d][s] so PV B-tiles are contiguous loads
__global__ void k_pack_v(const float* __restrict__ qkv, bf16* __restrict__ Vt) {
  int i = blockIdx.x * blockDim.x + threadIdx.x;       // TOKENS*8*64
  if (i >= TOKENS * NUM_KV_HEADS * HEAD_DIM) return;
  int d = i & 63; int t = i >> 6;
  int kvh = t & (NUM_KV_HEADS - 1); int tok = t >> 3;
  int b = tok / SEQ, s = tok % SEQ;
  float v = qkv[(size_t)tok * QKV_OUT + D_MODEL + NUM_KV_HEADS * HEAD_DIM + kvh * HEAD_DIM + d];
  Vt[((((size_t)(b * NUM_KV_HEADS + kvh)) * HEAD_DIM) + d) * SEQ + s] = (bf16)v;
}

// ---------------------------------------------------------------- flash attention
// Wave = 16 Q rows; block = 8 waves = 128 rows; grid (SEQ/128, H, B).
__global__ __launch_bounds__(256)
void k_flash(const bf16* __restrict__ Q, const bf16* __restrict__ Km,
             const bf16* __restrict__ Vt, bf16* __restrict__ O) {
  __shared__ __align__(16) bf16 smem[8][16][32];   // per-wave P round-trip tile (1 KB each)

  const int lane = threadIdx.x & 31;
  const int wave = threadIdx.x >> 5;
  const int b = blockIdx.z, h = blockIdx.y;
  const int kvh = h % NUM_KV_HEADS;              // jnp.tile -> h % 8
  const int q0 = blockIdx.x * 128 + wave * 16;

  const bf16* Qh = Q  + ((size_t)(b * NUM_HEADS    + h  ) * SEQ) * HEAD_DIM;
  const bf16* Kh = Km + ((size_t)(b * NUM_KV_HEADS + kvh) * SEQ) * HEAD_DIM;
  const bf16* Vh = Vt + ((size_t)(b * NUM_KV_HEADS + kvh) * HEAD_DIM) * SEQ;

  const int mrow = lane & 15;
  const int kb   = (lane < 16) ? 0 : 8;
  const int kb2  = (lane < 16) ? 0 : 16;

  // Q tile 16x64 as two A matrices (d=0..31, 32..63)
  v16bf aq[2];
#pragma unroll
  for (int t = 0; t < 2; t++) {
    const bf16* p = Qh + (size_t)(q0 + mrow) * HEAD_DIM + t * 32 + kb;
    v8bf lo = *(const v8bf*)p;
    v8bf hi = *(const v8bf*)(p + 16);
    aq[t] = CAT16(lo, hi);
  }

  v8f oacc[4] = {};
  float rm[8], rl[8];
#pragma unroll
  for (int v = 0; v < 8; v++) { rm[v] = -3.0e38f; rl[v] = 0.0f; }

  const int qlast = q0 + 15;
  for (int j0 = 0; j0 <= qlast; j0 += 32) {
    // scores 16x32 = two 16x16 C tiles, each from two K=32 WMMAs over head_dim
    v8f sc[2];
#pragma unroll
    for (int jt = 0; jt < 2; jt++) {
      const bf16* kp = Kh + (size_t)(j0 + jt * 16 + mrow) * HEAD_DIM;
      v16bf bk0 = CAT16(*(const v8bf*)(kp + kb), *(const v8bf*)(kp + kb + 16));
      v16bf bk1 = CAT16(*(const v8bf*)(kp + 32 + kb), *(const v8bf*)(kp + 32 + kb + 16));
      v8f z = {};
      z = __builtin_amdgcn_wmma_f32_16x16x32_bf16(false, aq[0], false, bk0, (short)0, z, false, false);
      z = __builtin_amdgcn_wmma_f32_16x16x32_bf16(false, aq[1], false, bk1, (short)0, z, false, false);
      sc[jt] = z;
    }
    // online softmax (per-row; rows live in 16-lane halves of each C vgpr)
#pragma unroll
    for (int v = 0; v < 8; v++) {
      int row = q0 + v + ((lane < 16) ? 0 : 8);
      float s0 = (j0 + mrow      <= row) ? sc[0][v] * 0.125f : -3.0e38f;
      float s1 = (j0 + 16 + mrow <= row) ? sc[1][v] * 0.125f : -3.0e38f;
      float mx = fmaxf(s0, s1);
      mx = fmaxf(mx, __shfl_xor(mx, 1, 16));
      mx = fmaxf(mx, __shfl_xor(mx, 2, 16));
      mx = fmaxf(mx, __shfl_xor(mx, 4, 16));
      mx = fmaxf(mx, __shfl_xor(mx, 8, 16));
      float mnew = fmaxf(rm[v], mx);
      float p0 = __expf(s0 - mnew), p1 = __expf(s1 - mnew);
      float rs = p0 + p1;
      rs += __shfl_xor(rs, 1, 16);
      rs += __shfl_xor(rs, 2, 16);
      rs += __shfl_xor(rs, 4, 16);
      rs += __shfl_xor(rs, 8, 16);
      float alpha = __expf(rm[v] - mnew);
      rl[v] = rl[v] * alpha + rs;
      rm[v] = mnew;
#pragma unroll
      for (int dt = 0; dt < 4; dt++) oacc[dt][v] *= alpha;
      int r = v + ((lane < 16) ? 0 : 8);
      smem[wave][r][mrow]      = (bf16)p0;
      smem[wave][r][16 + mrow] = (bf16)p1;
    }
    asm volatile("s_wait_dscnt 0" ::: "memory");   // C-layout stores visible before A-layout reads
    const bf16* pp = &smem[wave][mrow][0];
    v8bf plo = *(const v8bf*)(pp + kb);
    v8bf phi = *(const v8bf*)(pp + kb + 16);
    v16bf pa = CAT16(plo, phi);
    // O(16x64) += P(16x32) @ V(32x64); Vt rows give contiguous B-layout loads
#pragma unroll
    for (int dt = 0; dt < 4; dt++) {
      const bf16* vp = Vh + (size_t)(dt * 16 + mrow) * SEQ + j0 + kb;
      v16bf bv = CAT16(*(const v8bf*)vp, *(const v8bf*)(vp + 16));
      oacc[dt] = __builtin_amdgcn_wmma_f32_16x16x32_bf16(false, pa, false, bv,
                                                         (short)0, oacc[dt], false, false);
    }
  }
  // normalize, write O as bf16 in [b][s][h*64+d] (ready for out-proj GEMM)
#pragma unroll
  for (int dt = 0; dt < 4; dt++) {
    int d = dt * 16 + mrow;
#pragma unroll
    for (int v = 0; v < 8; v++) {
      int row = q0 + v + ((lane < 16) ? 0 : 8);
      float val = oacc[dt][v] / rl[v];
      O[((size_t)(b * SEQ + row)) * D_MODEL + h * HEAD_DIM + d] = (bf16)val;
    }
  }
}

// ---------------------------------------------------------------- launch
extern "C" void kernel_launch(void* const* d_in, const int* in_sizes, int n_in,
                              void* d_out, int out_size, void* d_ws, size_t ws_size,
                              hipStream_t stream) {
  (void)in_sizes; (void)n_in; (void)out_size; (void)ws_size;
  const float* x     = (const float*)d_in[0];
  const float* w_qkv = (const float*)d_in[1];
  const float* b_qkv = (const float*)d_in[2];
  const float* w_o   = (const float*)d_in[3];
  const float* b_o   = (const float*)d_in[4];
  float* out = (float*)d_out;

  char* wsp = (char*)d_ws;
  size_t off = 0;
  auto alloc = [&](size_t bytes) -> void* {
    void* p = wsp + off;
    off = (off + bytes + 255) & ~(size_t)255;
    return p;
  };
  bf16*  xbf   = (bf16*) alloc((size_t)TOKENS * D_MODEL * 2);
  bf16*  wqkvT = (bf16*) alloc((size_t)QKV_OUT * D_MODEL * 2);
  bf16*  woT   = (bf16*) alloc((size_t)D_MODEL * D_MODEL * 2);
  float* qkv   = (float*)alloc((size_t)TOKENS * QKV_OUT * 4);
  bf16*  Qbf   = (bf16*) alloc((size_t)BATCH * NUM_HEADS    * SEQ * HEAD_DIM * 2);
  bf16*  Kbf   = (bf16*) alloc((size_t)BATCH * NUM_KV_HEADS * SEQ * HEAD_DIM * 2);
  bf16*  Vt    = (bf16*) alloc((size_t)BATCH * NUM_KV_HEADS * SEQ * HEAD_DIM * 2);
  bf16*  Obf   = (bf16*) alloc((size_t)TOKENS * D_MODEL * 2);

  { int n = TOKENS * D_MODEL;
    k_f32_to_bf16<<<(n + 255) / 256, 256, 0, stream>>>(x, xbf, n); }
  { int n = D_MODEL * QKV_OUT;
    k_transpose_bf16<<<(n + 255) / 256, 256, 0, stream>>>(w_qkv, wqkvT, D_MODEL, QKV_OUT); }
  { int n = D_MODEL * D_MODEL;
    k_transpose_bf16<<<(n + 255) / 256, 256, 0, stream>>>(w_o, woT, D_MODEL, D_MODEL); }

  // QKV projection: [4096,2048] @ [2048,3072]
  k_gemm_bf16<<<dim3(QKV_OUT / 128, TOKENS / 128), 256, 0, stream>>>(
      xbf, wqkvT, b_qkv, qkv, TOKENS, QKV_OUT, D_MODEL);

  { int n = TOKENS * NUM_HEADS * 32;
    k_rope_q<<<(n + 255) / 256, 256, 0, stream>>>(qkv, Qbf); }
  { int n = TOKENS * NUM_KV_HEADS * 32;
    k_rope_k<<<(n + 255) / 256, 256, 0, stream>>>(qkv, Kbf); }
  { int n = TOKENS * NUM_KV_HEADS * HEAD_DIM;
    k_pack_v<<<(n + 255) / 256, 256, 0, stream>>>(qkv, Vt); }

  k_flash<<<dim3(SEQ / 128, NUM_HEADS, BATCH), 256, 0, stream>>>(Qbf, Kbf, Vt, Obf);

  // Output projection: [4096,2048] @ [2048,2048]
  k_gemm_bf16<<<dim3(D_MODEL / 128, TOKENS / 128), 256, 0, stream>>>(
      Obf, woT, b_o, out, TOKENS, D_MODEL, D_MODEL);
}
